// SpatialAttentionHead_47974784696833
// MI455X (gfx1250) — matrix operand order repaired and checked by reading, not compiled, not run on hardware
//
#include <hip/hip_runtime.h>

typedef _Float16 half_t;
typedef __attribute__((ext_vector_type(16))) _Float16 v16h;
typedef __attribute__((ext_vector_type(8)))  _Float16 v8h;
typedef __attribute__((ext_vector_type(8)))  float    v8f;

#define NN 8192
#define DD 256
#define FF 64
#define CHUNK 32
#define NCH (NN / CHUNK)
#define KPAD 72   // halves per Klds row (32 keys x 64 d, padded)
#define VPAD 40   // halves per VTlds row (64 f x 32 keys, padded)

union H2U { half_t h[2]; unsigned int u; };
union V16U { v16h v; unsigned int u[8]; };

// CDNA5 async global->LDS copy (ASYNCcnt-tracked), 16 bytes per lane.
// Generic pointer to __shared__ has the LDS byte offset in its low 32 bits.
__device__ __forceinline__ void async_copy16(const half_t* g, half_t* l) {
    unsigned lds = (unsigned)(unsigned long long)(void*)l;
    asm volatile("global_load_async_to_lds_b128 %0, %1, off"
                 :: "v"(lds), "v"(g) : "memory");
}
__device__ __forceinline__ void wait_async0() {
    asm volatile("s_wait_asynccnt 0x0" ::: "memory");
}

// ---------------------------------------------------------------------------
// Projection kernel: one block (256 thr, 8 waves) per node i.
//   qh = (x[i]·Wq[i]^T + bq[i]) * 0.125  (f16, softmax scale pre-folded)
//   kh = x[i]·Wk^T + bk                   (f16, row-major N x F)
//   vT = (x[i]·Wv^T + bv) transposed      (f16, F x N)
// Bandwidth-bound on the 512MB Wq stream. Each 8-lane group owns one output
// row (lane reads a contiguous 128B slice of the weight row), so the cross-
// lane reduction is only a 3-level butterfly.
// ---------------------------------------------------------------------------
__global__ __launch_bounds__(256) void proj_kernel(
    const float* __restrict__ x,
    const float* __restrict__ Wk, const float* __restrict__ bk,
    const float* __restrict__ Wv, const float* __restrict__ bv,
    const float* __restrict__ Wq, const float* __restrict__ bq,
    half_t* __restrict__ qh, half_t* __restrict__ kh, half_t* __restrict__ vT)
{
    __shared__ float xs[DD];
    const int i   = blockIdx.x;
    const int tid = threadIdx.x;
    xs[tid] = x[(size_t)i * DD + tid];
    __syncthreads();

    const int lane = tid & 31;
    const int wave = tid >> 5;
    const int sub  = lane & 7;    // position within 8-lane group
    const int g    = lane >> 3;   // group -> row within batch of 4

    float xv[32];
#pragma unroll
    for (int j = 0; j < 32; ++j) xv[j] = xs[sub * 32 + j];

#pragma unroll
    for (int round = 0; round < 2; ++round) {
        const int f = wave * 8 + round * 4 + g;
        const float* qrow = Wq + (size_t)i * FF * DD + (size_t)f * DD + sub * 32;
        const float* krow = Wk + (size_t)f * DD + sub * 32;
        const float* vrow = Wv + (size_t)f * DD + sub * 32;
        float aq = 0.f, ak = 0.f, av = 0.f;
#pragma unroll
        for (int j = 0; j < 32; ++j) {
            aq += qrow[j] * xv[j];
            ak += krow[j] * xv[j];
            av += vrow[j] * xv[j];
        }
#pragma unroll
        for (int m = 1; m < 8; m <<= 1) {
            aq += __shfl_xor(aq, m, 32);
            ak += __shfl_xor(ak, m, 32);
            av += __shfl_xor(av, m, 32);
        }
        if (sub == 0) {
            qh[(size_t)i * FF + f] = (half_t)((aq + bq[(size_t)i * FF + f]) * 0.125f);
            kh[(size_t)i * FF + f] = (half_t)(ak + bk[f]);
            vT[(size_t)f * NN + i] = (half_t)(av + bv[f]);
        }
    }
}

// ---------------------------------------------------------------------------
// Flash attention: 1 wave per 16-query tile, 4 waves/block, 128 blocks.
// Double-buffered async global->LDS staging (ASYNCcnt) of K and V^T chunks;
// one barrier per chunk. Computes S^T tiles (A = K-tile, B = Q) so each lane
// owns one query column: online softmax is 8 local ops + one shfl_xor(16),
// and P^T re-lays into B-matrix form with 8 lane-pair shuffles.
// PV uses A = V^T tiles (contiguous b128 LDS loads via pre-transposed vT).
// 8 x v_wmma_f32_16x16x32_f16 per 32-key chunk.
// ---------------------------------------------------------------------------
__global__ __launch_bounds__(128) void attn_kernel(
    const half_t* __restrict__ qh, const half_t* __restrict__ kh,
    const half_t* __restrict__ vT, float* __restrict__ out)
{
    __shared__ half_t Klds[2][CHUNK * KPAD];   // [buf][key][d], padded rows
    __shared__ half_t VTlds[2][FF * VPAD];     // [buf][f][key], padded rows

    const int tid  = threadIdx.x;
    const int lane = tid & 31;
    const int wave = tid >> 5;
    const int h    = lane >> 4;     // wave half (K-dim split for A/B layouts)
    const int n    = lane & 15;     // query column owned by this lane
    const int q0   = blockIdx.x * 64 + wave * 16;

    // Per-thread staging offsets (constant over the chunk loop):
    // K chunk = 32 rows x 64 halves, 2 x b128 per thread.
    const int krow = tid >> 3, kcol = tid & 7;
    const int kG0  = krow * FF + kcol * 8;             // halves into chunk
    const int kL0  = krow * KPAD + kcol * 8;
    // VT chunk = 64 rows x 32 halves, 2 x b128 per thread.
    const int vrow = tid >> 2, vcol = tid & 3;
    const size_t vG0 = (size_t)vrow * NN + vcol * 8;   // halves into vT (+ c*32)
    const int vL0  = vrow * VPAD + vcol * 8;

    auto stage = [&](int c, int buf) {
        const half_t* kbase = kh + (size_t)c * CHUNK * FF;
        const half_t* vbase = vT + (size_t)c * CHUNK;
        async_copy16(kbase + kG0,           &Klds[buf][kL0]);
        async_copy16(kbase + kG0 + 16 * FF, &Klds[buf][kL0 + 16 * KPAD]);
        async_copy16(vbase + vG0,                    &VTlds[buf][vL0]);
        async_copy16(vbase + vG0 + (size_t)32 * NN,  &VTlds[buf][vL0 + 32 * VPAD]);
    };

    // Q tile in B-matrix layout: lane holds query q0+n, d = tile*32 + h*16 + (0..15)
    v16h bq0, bq1;
    {
        const half_t* p = qh + (size_t)(q0 + n) * FF + h * 16;
        bq0 = *(const v16h*)(p);
        bq1 = *(const v16h*)(p + 32);
    }

    v8f acc[4];
#pragma unroll
    for (int t = 0; t < 4; ++t) acc[t] = (v8f)(0.f);
    float run_max = -1e30f, run_sum = 0.f;

    stage(0, 0);   // prologue: chunk 0 -> buffer 0

    for (int c = 0; c < NCH; ++c) {
        const int buf = c & 1;
        wait_async0();        // my async writes into buf are committed
        __syncthreads();      // everyone's are; prev compute on buf^1 done
        if (c + 1 < NCH) stage(c + 1, buf ^ 1);

        // ---- S^T tiles: st[g][r] = score(query n, key g*16 + r + 8h) ----
        v8f st[2];
#pragma unroll
        for (int g = 0; g < 2; ++g) {
            const half_t* kb = &Klds[buf][(g * 16 + n) * KPAD];  // A row = key
            v8h lo0 = *(const v8h*)(kb + h * 16);
            v8h hi0 = *(const v8h*)(kb + 8 + h * 16);
            v8h lo1 = *(const v8h*)(kb + 32 + h * 16);
            v8h hi1 = *(const v8h*)(kb + 40 + h * 16);
            v16h a0 = __builtin_shufflevector(lo0, hi0, 0,1,2,3,4,5,6,7,8,9,10,11,12,13,14,15);
            v16h a1 = __builtin_shufflevector(lo1, hi1, 0,1,2,3,4,5,6,7,8,9,10,11,12,13,14,15);
            v8f sz = (v8f)(0.f);
            sz = __builtin_amdgcn_wmma_f32_16x16x32_f16(false, a0, false, bq0, (short)0, sz, false, false);
            sz = __builtin_amdgcn_wmma_f32_16x16x32_f16(false, a1, false, bq1, (short)0, sz, false, false);
            st[g] = sz;
        }

        // ---- online softmax over this 32-key chunk (per query = per lane) ----
        float cmax = -1e30f;
#pragma unroll
        for (int g = 0; g < 2; ++g)
#pragma unroll
            for (int r = 0; r < 8; ++r) cmax = fmaxf(cmax, st[g][r]);
        cmax = fmaxf(cmax, __shfl_xor(cmax, 16, 32));
        const float nmax = fmaxf(run_max, cmax);
        const float corr = __expf(run_max - nmax);
        float p[2][8];
        float csum = 0.f;
#pragma unroll
        for (int g = 0; g < 2; ++g)
#pragma unroll
            for (int r = 0; r < 8; ++r) {
                float e = __expf(st[g][r] - nmax);
                p[g][r] = e;
                csum += e;
            }
        csum += __shfl_xor(csum, 16, 32);
        run_sum = run_sum * corr + csum;
        run_max = nmax;
#pragma unroll
        for (int t = 0; t < 4; ++t)
#pragma unroll
            for (int r = 0; r < 8; ++r) acc[t][r] *= corr;

        // ---- P^T into B-matrix layout: lane needs keys h*16+(0..15) of its
        //      own query; half come from the lane^16 partner via shfl_xor ----
        unsigned int pk[2][4], ex[2][4];
#pragma unroll
        for (int g = 0; g < 2; ++g)
#pragma unroll
            for (int r = 0; r < 4; ++r) {
                H2U u;
                u.h[0] = (half_t)p[g][2 * r];
                u.h[1] = (half_t)p[g][2 * r + 1];
                pk[g][r] = u.u;
                ex[g][r] = (unsigned int)__shfl_xor((int)u.u, 16, 32);
            }
        V16U bp;
#pragma unroll
        for (int r = 0; r < 4; ++r) bp.u[r]     = h ? ex[1][r] : pk[0][r];
#pragma unroll
        for (int r = 0; r < 4; ++r) bp.u[4 + r] = h ? pk[1][r] : ex[0][r];

        // ---- PV: out^T tile t (features t*16..t*16+15), A = V^T tile ----
#pragma unroll
        for (int t = 0; t < 4; ++t) {
            const half_t* vb = &VTlds[buf][(t * 16 + n) * VPAD];  // A row = feature
            v8h lo = *(const v8h*)(vb + h * 16);
            v8h hi = *(const v8h*)(vb + 8 + h * 16);
            v16h av = __builtin_shufflevector(lo, hi, 0,1,2,3,4,5,6,7,8,9,10,11,12,13,14,15);
            acc[t] = __builtin_amdgcn_wmma_f32_16x16x32_f16(false, av, false, bp.v, (short)0, acc[t], false, false);
        }
    }

    // ---- normalize and write: lane owns query q0+n; acc[t][r] = f t*16+r+8h ----
    const float inv = 1.0f / run_sum;
#pragma unroll
    for (int t = 0; t < 4; ++t) {
        v8f o = acc[t] * inv;
        *(v8f*)(out + (size_t)(q0 + n) * FF + t * 16 + 8 * h) = o;
    }
}

extern "C" void kernel_launch(void* const* d_in, const int* in_sizes, int n_in,
                              void* d_out, int out_size, void* d_ws, size_t ws_size,
                              hipStream_t stream) {
    (void)in_sizes; (void)n_in; (void)out_size; (void)ws_size;
    const float* x  = (const float*)d_in[0];
    const float* Wk = (const float*)d_in[1];
    const float* bk = (const float*)d_in[2];
    const float* Wv = (const float*)d_in[3];
    const float* bv = (const float*)d_in[4];
    const float* Wq = (const float*)d_in[5];
    const float* bq = (const float*)d_in[6];
    float* out = (float*)d_out;

    half_t* qh = (half_t*)d_ws;            // N*F halves = 1MB
    half_t* kh = qh + (size_t)NN * FF;     // 1MB
    half_t* vT = kh + (size_t)NN * FF;     // 1MB (F x N, transposed V)

    proj_kernel<<<NN, 256, 0, stream>>>(x, Wk, bk, Wv, bv, Wq, bq, qh, kh, vT);
    attn_kernel<<<NN / 64, 128, 0, stream>>>(qh, kh, vT, out);
}
